// SelfAttention_89678917140885
// MI455X (gfx1250) — compile-verified, matrix operand order checked by reading, not run on hardware
//
#include <hip/hip_runtime.h>
#include <hip/hip_bf16.h>

#define B_  4
#define S_  2048
#define H_  16
#define DH_ 64
#define DM_ 1024
#define SC_LD 2052                    // score row stride (floats); 2052 % 64 == 4
                                      // -> 16 WMMA rows hit 16 distinct LDS banks
#define QSTAGE_OFF  (32 * SC_LD + 256)            // floats
#define ATTN_LDS_BYTES (QSTAGE_OFF * 4 + 4096)    // + 32x64 bf16 q stage

typedef __attribute__((ext_vector_type(16))) __bf16       v16bf;
typedef __attribute__((ext_vector_type(8)))  __bf16       v8bf;
typedef __attribute__((ext_vector_type(8)))  float        v8f;
typedef __attribute__((ext_vector_type(8)))  unsigned int v8u;

static __device__ inline v8f wmma_bf16(v16bf a, v16bf b, v8f c) {
  // (neg_a, A, neg_b, B, c_mod, C, reuse_a, reuse_b)
  return __builtin_amdgcn_wmma_f32_16x16x32_bf16(false, a, false, b, (short)0, c,
                                                 false, false);
}

// 16x32 bf16 operand tile (A rows = M, pre-transposed B rows = N), row-major,
// leading dim ld.  ISA 7.12.2: lane L -> row L&15; elems 0..7 = K hb..hb+7,
// 8..15 = K hb+16..hb+23, hb = 8 for lanes 16..31.
static __device__ inline v16bf load_tile_bf16(const __bf16* base, int ld) {
  const int L  = threadIdx.x & 31;
  const int r  = L & 15;
  const int hb = (L & 16) ? 8 : 0;
  const __bf16* p = base + (size_t)r * ld + hb;
  uint4 g0 = *(const uint4*)(p);
  uint4 g1 = *(const uint4*)(p + 16);
  v8u u;
  u[0] = g0.x; u[1] = g0.y; u[2] = g0.z; u[3] = g0.w;
  u[4] = g1.x; u[5] = g1.y; u[6] = g1.z; u[7] = g1.w;
  return __builtin_bit_cast(v16bf, u);
}

// ---------------------------------------------------------------- prep kernels
__global__ __launch_bounds__(256) void k_convert_x(const float* __restrict__ x,
                                                   __bf16* __restrict__ xb, int n) {
  int i = blockIdx.x * blockDim.x + threadIdx.x;
  int stride = gridDim.x * blockDim.x;
  for (; i < n; i += stride) xb[i] = (__bf16)x[i];
}

// W[h][d][n] (f32) -> Wt[h][n][d] (bf16) for Q,K,V
__global__ __launch_bounds__(256) void k_transpose_wqkv(
    const float* __restrict__ Wq, const float* __restrict__ Wk,
    const float* __restrict__ Wv, __bf16* __restrict__ WqT,
    __bf16* __restrict__ WkT, __bf16* __restrict__ WvT) {
  const int n = H_ * DM_ * DH_;
  int i = blockIdx.x * blockDim.x + threadIdx.x;
  int stride = gridDim.x * blockDim.x;
  for (; i < n; i += stride) {
    int h   = i / (DM_ * DH_);
    int rem = i - h * (DM_ * DH_);
    int d   = rem / DH_;
    int nn  = rem - d * DH_;
    int dst = (h * DH_ + nn) * DM_ + d;
    WqT[dst] = (__bf16)Wq[i];
    WkT[dst] = (__bf16)Wk[i];
    WvT[dst] = (__bf16)Wv[i];
  }
}

// W_O[h][k][dm] (f32) -> WoT[dm][h*64+k] (bf16)
__global__ __launch_bounds__(256) void k_transpose_wo(const float* __restrict__ Wo,
                                                      __bf16* __restrict__ WoT) {
  const int n = H_ * DH_ * DM_;
  int i = blockIdx.x * blockDim.x + threadIdx.x;
  int stride = gridDim.x * blockDim.x;
  for (; i < n; i += stride) {
    int hk = i / DM_;
    int dm = i - hk * DM_;
    WoT[(size_t)dm * (H_ * DH_) + hk] = (__bf16)Wo[i];
  }
}

// ------------------------------------------------------------------- QKV GEMM
// wave = one 16-token tile x one head; computes q,k,v (16x64 each).
__global__ __launch_bounds__(256) void k_qkv(
    const __bf16* __restrict__ xb, const __bf16* __restrict__ WqT,
    const __bf16* __restrict__ WkT, const __bf16* __restrict__ WvT,
    const float* __restrict__ bQ, const float* __restrict__ bK,
    const float* __restrict__ bV, __bf16* __restrict__ qb,
    __bf16* __restrict__ kb, __bf16* __restrict__ vT) {
  const int wave = (blockIdx.x << 3) + (threadIdx.x >> 5);
  const int h    = wave & (H_ - 1);
  const int tt   = wave >> 4;  // token tile 0..511
  const int L    = threadIdx.x & 31;
  const int ln   = L & 15;
  const int hb8  = (L & 16) ? 8 : 0;

  v8f accQ[4], accK[4], accV[4];
#pragma unroll
  for (int t = 0; t < 4; ++t) { accQ[t] = {}; accK[t] = {}; accV[t] = {}; }

  const __bf16* abase = xb + (size_t)tt * 16 * DM_;
  const __bf16* wq = WqT + (size_t)h * DH_ * DM_;
  const __bf16* wk = WkT + (size_t)h * DH_ * DM_;
  const __bf16* wv = WvT + (size_t)h * DH_ * DM_;

  for (int kc = 0; kc < DM_; kc += 32) {
    v16bf a = load_tile_bf16(abase + kc, DM_);
#pragma unroll
    for (int t = 0; t < 4; ++t) {
      accQ[t] = wmma_bf16(a, load_tile_bf16(wq + (size_t)(t * 16) * DM_ + kc, DM_), accQ[t]);
      accK[t] = wmma_bf16(a, load_tile_bf16(wk + (size_t)(t * 16) * DM_ + kc, DM_), accK[t]);
      accV[t] = wmma_bf16(a, load_tile_bf16(wv + (size_t)(t * 16) * DM_ + kc, DM_), accV[t]);
    }
  }

  const int b  = (tt * 16) >> 11;  // token tiles never cross a batch
  const int s0 = (tt * 16) & (S_ - 1);
#pragma unroll
  for (int t = 0; t < 4; ++t) {
    const int   n  = t * 16 + ln;
    const float vq = bQ[h * DH_ + n];
    const float vk = bK[h * DH_ + n];
    const float vv = bV[h * DH_ + n];
    // q,k: [b][h][s][64]
    size_t qi = (((size_t)(b * H_ + h)) * S_ + s0 + hb8) * DH_ + n;
#pragma unroll
    for (int r = 0; r < 8; ++r) {
      qb[qi + (size_t)r * DH_] = (__bf16)(accQ[t][r] + vq);
      kb[qi + (size_t)r * DH_] = (__bf16)(accK[t][r] + vk);
    }
    // vT: [b][h][64][S] -> contiguous 8-token store per lane
    v8bf pk;
#pragma unroll
    for (int r = 0; r < 8; ++r) pk[r] = (__bf16)(accV[t][r] + vv);
    size_t vi = (((size_t)(b * H_ + h)) * DH_ + n) * S_ + s0 + hb8;
    *(v8bf*)(vT + vi) = pk;
  }
}

// ---------------------------------------------------- fused scores+softmax+z
// One workgroup (8 waves) per (b*h, 32-row q block).  Score block lives in LDS
// (32 x 2052 f32 = 262.7 KB of the 320 KB WGP LDS); q block is staged into LDS
// with global_load_async_to_lds_b128; pattern is written to global exactly
// once (float4 stores); z = P@V reads probabilities straight from LDS.
__global__ __launch_bounds__(256) void k_attn(const __bf16* __restrict__ qb,
                                              const __bf16* __restrict__ kb,
                                              const __bf16* __restrict__ vT,
                                              float* __restrict__ patt,
                                              __bf16* __restrict__ zb) {
  extern __shared__ float smem[];
  float*  sc      = smem;                        // [32][SC_LD]
  float*  red     = smem + 32 * SC_LD;           // [256]
  __bf16* qs_lds  = (__bf16*)(smem + QSTAGE_OFF);  // [32][64] staged q block

  const int qb32  = blockIdx.x & 63;   // q block within sequence
  const int bh    = blockIdx.x >> 6;   // 0..63
  const int qbase = qb32 * 32;
  const int kend  = qbase + 32;        // keys 0..kend-1 are causally live
  const int nkt   = kend >> 4;         // live 16-wide k tiles
  const int w     = threadIdx.x >> 5;
  const int L     = threadIdx.x & 31;
  const int ln    = L & 15;
  const int hb8   = (L & 16) ? 8 : 0;

  // ---- phase 0: async-stage the 32x64 bf16 q block into LDS ---------------
  {
    const __bf16* gsrc =
        qb + ((size_t)bh * S_ + qbase) * DH_ + threadIdx.x * 8;  // 16B per lane
    unsigned ldsoff = (unsigned)(uintptr_t)smem + QSTAGE_OFF * 4 +
                      threadIdx.x * 16;
    asm volatile("global_load_async_to_lds_b128 %0, %1, off"
                 :
                 : "v"(ldsoff), "v"(gsrc)
                 : "memory");
    asm volatile("s_wait_asynccnt 0x0" ::: "memory");
  }
  __syncthreads();

  // ---- phase 1: scores -> LDS (masked, /sqrt(64)) -------------------------
  for (int j = w; j < 2 * nkt; j += 8) {
    const int qs = j & 1;        // q subtile (rows qs*16..)
    const int kt = j >> 1;       // k tile
    const __bf16* qrow = qs_lds + qs * 16 * DH_;
    v16bf a0 = load_tile_bf16(qrow, DH_);
    v16bf a1 = load_tile_bf16(qrow + 32, DH_);
    const __bf16* krow = kb + ((size_t)bh * S_ + kt * 16) * DH_;
    v8f acc = {};
    acc = wmma_bf16(a0, load_tile_bf16(krow, DH_), acc);
    acc = wmma_bf16(a1, load_tile_bf16(krow + 32, DH_), acc);
    const int kcol = kt * 16 + ln;
#pragma unroll
    for (int r = 0; r < 8; ++r) {
      const int qg = qbase + qs * 16 + r + hb8;
      sc[(qs * 16 + r + hb8) * SC_LD + kcol] =
          (kcol > qg) ? -100000.0f : acc[r] * 0.125f;
    }
  }
  __syncthreads();

  // ---- phase 2: rowwise softmax (float4); write pattern once --------------
  {
    const int row = threadIdx.x >> 3;  // 0..31
    const int g   = threadIdx.x & 7;   // 8 threads per row
    float4* srow4 = (float4*)(sc + row * SC_LD);
    const int nl4 = kend >> 2;         // live float4s per row
    float m = -1e30f;
    for (int c = g; c < nl4; c += 8) {
      float4 v = srow4[c];
      m = fmaxf(m, fmaxf(fmaxf(v.x, v.y), fmaxf(v.z, v.w)));
    }
    red[threadIdx.x] = m;
    __syncthreads();
    if (g < 4) red[threadIdx.x] = fmaxf(red[threadIdx.x], red[threadIdx.x + 4]);
    __syncthreads();
    if (g < 2) red[threadIdx.x] = fmaxf(red[threadIdx.x], red[threadIdx.x + 2]);
    __syncthreads();
    if (g < 1) red[threadIdx.x] = fmaxf(red[threadIdx.x], red[threadIdx.x + 1]);
    __syncthreads();
    m = red[row * 8];
    __syncthreads();
    float s = 0.f;
    for (int c = g; c < nl4; c += 8) {
      float4 v = srow4[c];
      v.x = __expf(v.x - m);
      v.y = __expf(v.y - m);
      v.z = __expf(v.z - m);
      v.w = __expf(v.w - m);
      s += (v.x + v.y) + (v.z + v.w);
      srow4[c] = v;
    }
    red[threadIdx.x] = s;
    __syncthreads();
    if (g < 4) red[threadIdx.x] += red[threadIdx.x + 4];
    __syncthreads();
    if (g < 2) red[threadIdx.x] += red[threadIdx.x + 2];
    __syncthreads();
    if (g < 1) red[threadIdx.x] += red[threadIdx.x + 1];
    __syncthreads();
    const float inv = 1.0f / red[row * 8];
    float4* prow4 = (float4*)(patt + ((size_t)bh * S_ + qbase + row) * S_);
    for (int c = g; c < nl4; c += 8) {
      float4 v = srow4[c];
      v.x *= inv; v.y *= inv; v.z *= inv; v.w *= inv;
      srow4[c] = v;   // probabilities stay in LDS for phase 3
      prow4[c] = v;   // 128B-coalesced pattern write
    }
    const float4 zero4 = make_float4(0.f, 0.f, 0.f, 0.f);
    for (int c = nl4 + g; c < (S_ >> 2); c += 8) prow4[c] = zero4;  // masked tail
  }
  __syncthreads();

  // ---- phase 3: z = P @ V (A tiles from LDS, causally clipped K loop) -----
  {
    const int qs = w >> 2;  // 0..1
    const int nt = w & 3;   // 0..3
    const int hb = hb8;
    v8f acc = {};
    const __bf16* vbase = vT + ((size_t)bh * DH_ + nt * 16) * S_;
    const int npc = kend >> 5;
    for (int pc = 0; pc < npc; ++pc) {
      const float* p = sc + (qs * 16 + ln) * SC_LD + pc * 32 + hb;
      v16bf a;
#pragma unroll
      for (int i = 0; i < 8; ++i) {
        a[i]     = (__bf16)p[i];
        a[i + 8] = (__bf16)p[i + 16];
      }
      acc = wmma_bf16(a, load_tile_bf16(vbase + pc * 32, S_), acc);
    }
    const int    b    = bh >> 4;
    const int    h    = bh & 15;
    const size_t tok0 = (size_t)b * S_ + qbase + qs * 16 + hb8;
    const int    n    = h * DH_ + nt * 16 + ln;
#pragma unroll
    for (int r = 0; r < 8; ++r)
      zb[(tok0 + r) * (H_ * DH_) + n] = (__bf16)acc[r];
  }
}

// ------------------------------------------------------------ out projection
__global__ __launch_bounds__(256) void k_oproj(const __bf16* __restrict__ zb,
                                               const __bf16* __restrict__ WoT,
                                               const float* __restrict__ bO,
                                               float* __restrict__ out) {
  const int wave = (blockIdx.x << 3) + (threadIdx.x >> 5);
  const int nt   = wave & 63;
  const int tt   = wave >> 6;  // 0..511
  const int L    = threadIdx.x & 31;
  const int ln   = L & 15;
  const int hb8  = (L & 16) ? 8 : 0;

  v8f acc = {};
  const __bf16* abase = zb + (size_t)tt * 16 * DM_;
  const __bf16* bbase = WoT + (size_t)nt * 16 * DM_;
  for (int kc = 0; kc < DM_; kc += 32) {
    if (kc + 32 < DM_) __builtin_prefetch(abase + kc + 32, 0, 0);  // global_prefetch_b8
    acc = wmma_bf16(load_tile_bf16(abase + kc, DM_),
                    load_tile_bf16(bbase + kc, DM_), acc);
  }

  const int    n    = nt * 16 + ln;
  const float  bias = bO[n];
  const size_t tok0 = (size_t)tt * 16 + hb8;
#pragma unroll
  for (int r = 0; r < 8; ++r) out[(tok0 + r) * DM_ + n] = acc[r] + bias;
}

// ---------------------------------------------------------------------- host
extern "C" void kernel_launch(void* const* d_in, const int* in_sizes, int n_in,
                              void* d_out, int out_size, void* d_ws, size_t ws_size,
                              hipStream_t stream) {
  const float* x  = (const float*)d_in[0];
  const float* Wq = (const float*)d_in[1];
  const float* bQ = (const float*)d_in[2];
  const float* Wk = (const float*)d_in[3];
  const float* bK = (const float*)d_in[4];
  const float* Wv = (const float*)d_in[5];
  const float* bV = (const float*)d_in[6];
  const float* Wo = (const float*)d_in[7];
  const float* bO = (const float*)d_in[8];

  float* attn_out = (float*)d_out;                          // 8,388,608 f32
  float* pattern  = (float*)d_out + (size_t)B_ * S_ * DM_;  // 268,435,456 f32

  // workspace carve-up (256B aligned)
  char*  ws  = (char*)d_ws;
  size_t off = 0;
  auto carve = [&](size_t bytes) {
    char* p = ws + off;
    off += (bytes + 255) & ~(size_t)255;
    return p;
  };
  const size_t XB = (size_t)B_ * S_ * DM_ * 2;        // 16 MB
  const size_t WT = (size_t)H_ * DH_ * DM_ * 2;       // 2 MB
  const size_t QB = (size_t)B_ * H_ * S_ * DH_ * 2;   // 16 MB
  __bf16* xb  = (__bf16*)carve(XB);
  __bf16* WqT = (__bf16*)carve(WT);
  __bf16* WkT = (__bf16*)carve(WT);
  __bf16* WvT = (__bf16*)carve(WT);
  __bf16* WoT = (__bf16*)carve(WT);
  __bf16* qb  = (__bf16*)carve(QB);
  __bf16* kb  = (__bf16*)carve(QB);
  __bf16* vT  = (__bf16*)carve(QB);
  __bf16* zb  = (__bf16*)carve(QB);
  (void)ws_size; (void)in_sizes; (void)n_in; (void)out_size;

  static bool attr_set = []() {
    hipFuncSetAttribute(reinterpret_cast<const void*>(k_attn),
                        hipFuncAttributeMaxDynamicSharedMemorySize,
                        ATTN_LDS_BYTES);
    return true;
  }();
  (void)attr_set;

  k_convert_x<<<8192, 256, 0, stream>>>(x, xb, B_ * S_ * DM_);
  k_transpose_wqkv<<<1024, 256, 0, stream>>>(Wq, Wk, Wv, WqT, WkT, WvT);
  k_transpose_wo<<<1024, 256, 0, stream>>>(Wo, WoT);
  k_qkv<<<1024, 256, 0, stream>>>(xb, WqT, WkT, WvT, bQ, bK, bV, qb, kb, vT);
  k_attn<<<B_ * H_ * (S_ / 32), 256, ATTN_LDS_BYTES, stream>>>(qb, kb, vT,
                                                              pattern, zb);
  k_oproj<<<4096, 256, 0, stream>>>(zb, WoT, bO, attn_out);
}
